// DCellNN_71133248356980
// MI455X (gfx1250) — matrix operand order repaired and checked by reading, not compiled, not run on hardware
//
#include <hip/hip_runtime.h>
#include <math.h>

// ---------------- problem constants ----------------
#define BATCH 4096
#define GDIM  3008          // K of the big GEMM (3008 = 94*32)
#define T0N   512
#define T1N   64
#define CCH   8
#define KGN   16
#define HN    6
#define TGN   577           // T0+T1+1
#define NFULL 9232          // TGN*KGN  (N of the big GEMM)
#define NPAD  9344          // 73*128, zero-padded weight rows
#define OUTW  578           // T0+T1+2 output columns

typedef __attribute__((ext_vector_type(8)))  float  v8f;
typedef __attribute__((ext_vector_type(8)))  __bf16 v8bf;
typedef __attribute__((ext_vector_type(16))) __bf16 v16bf;

__device__ __forceinline__ unsigned short f2bf(float f) {
  union { float f; unsigned int u; } x; x.f = f;
  unsigned int u = x.u;
  unsigned int r = u + 0x7FFFu + ((u >> 16) & 1u);   // round-to-nearest-even
  return (unsigned short)(r >> 16);
}

// Convert f32 -> bf16, zero-padding rows >= validRows (for the padded weight matrix).
__global__ void cvt_bf16_pad(const float* __restrict__ src, unsigned short* __restrict__ dst,
                             long long validRows, long long totalRows, int cols) {
  long long i = (long long)blockIdx.x * blockDim.x + threadIdx.x;
  if (i >= totalRows * (long long)cols) return;
  long long row = i / cols;
  float v = (row < validRows) ? src[i] : 0.0f;
  dst[i] = f2bf(v);
}

// ---------------- big GEMM: gene_out = x @ Wg^T + bg ----------------
// C[4096, NPAD] = A[4096, 3008](bf16) x W[NPAD, 3008](bf16, row = N, K-contiguous)
// Block: 8 waves as 4(M) x 2(N), tile 64M x 128N.
// B tile (128 cols x 32 K = 8KB bf16) is staged in LDS via global_load_async_to_lds_b128
// (ASYNCcnt, double-buffered); each lane then reads its 32-B fragment with aligned DS loads.
// A fragments are loaded directly (only 2-way reuse).
__global__ __launch_bounds__(256) void gemm_gene_wmma(
    const unsigned short* __restrict__ Au,
    const unsigned short* __restrict__ Wu,
    const float* __restrict__ bias,
    float* __restrict__ Cout)
{
  __shared__ unsigned short bsm[2][128 * 32];   // 2 x 8KB B-tile buffers

  const int tid   = threadIdx.x;
  const int lane  = tid & 31;
  const int wave  = tid >> 5;
  const int waveM = wave & 3;
  const int waveN = wave >> 2;
  const int rowBase = blockIdx.y * 64 + waveM * 16;
  const int colBase = blockIdx.x * 128;          // block-level N base
  const int colW    = colBase + waveN * 64;      // this wave's N strip
  const int r  = lane & 15;       // M (A) / N (B) index within the 16x16 tile
  const int hi = lane >> 4;       // lane-half selects K sub-range: K 16*hi..16*hi+15 (contiguous)

  // ---- async B staging assignment: thread -> (local col, 32B half of 64B K-row) ----
  const int bc = tid >> 1;        // 0..127 local column
  const int bh = tid & 1;         // which 32B half
  const unsigned short* gB = Wu + (size_t)(colBase + bc) * GDIM + bh * 16;

  const __bf16* A = reinterpret_cast<const __bf16*>(Au);
  const __bf16* arow = A + (size_t)(rowBase + r) * GDIM + hi * 16;

  auto issueB = [&](int k0, int sel) {
    const unsigned short* gsrc = gB + k0;                          // 32 contiguous bytes
    unsigned ldsoff = (unsigned)(size_t)&bsm[sel][bc * 32 + bh * 16];
    // inst offset applies to both LDS and global address
    asm volatile("global_load_async_to_lds_b128 %0, %1, off"
                 :: "v"(ldsoff), "v"(gsrc) : "memory");
    asm volatile("global_load_async_to_lds_b128 %0, %1, off offset:16"
                 :: "v"(ldsoff), "v"(gsrc) : "memory");
  };

  v8f acc[4] = {};
  const int NSTEPS = GDIM / 32;   // 94
  issueB(0, 0);
  for (int it = 0; it < NSTEPS; ++it) {
    const int k0 = it * 32;
    if (it + 1 < NSTEPS) {
      issueB(k0 + 32, (it + 1) & 1);
      asm volatile("s_wait_asynccnt 0x2" ::: "memory");   // current stage landed
    } else {
      asm volatile("s_wait_asynccnt 0x0" ::: "memory");
    }
    __syncthreads();              // stage visible to all waves

    // A 16x32 bf16 fragment: lanes 0-15 hold K {0..15}; lanes 16-31 hold {16..31}
    v8bf alo = *reinterpret_cast<const v8bf*>(arow + k0);
    v8bf ahi = *reinterpret_cast<const v8bf*>(arow + k0 + 8);
    v16bf a = __builtin_shufflevector(alo, ahi, 0,1,2,3,4,5,6,7,8,9,10,11,12,13,14,15);
    __builtin_prefetch(arow + k0 + 256, 0, 3);

    // B 32x16 fragment from LDS: lane holds column N=r, 16 contiguous K at (hi*16)
    const unsigned short* bbase = &bsm[it & 1][(waveN * 64 + r) * 32 + hi * 16];
#pragma unroll
    for (int j = 0; j < 4; ++j) {
      v16bf b = *reinterpret_cast<const v16bf*>(bbase + j * 16 * 32);
      acc[j] = __builtin_amdgcn_wmma_f32_16x16x32_bf16(
          false, a, false, b, (short)0, acc[j], false, false);
    }
    __syncthreads();              // all reads done before this buffer is overwritten
  }

  // Epilogue: C/D layout — VGPR rr, lanes 0-15 -> M=rr, lanes 16-31 -> M=rr+8; N = lane%16
#pragma unroll
  for (int j = 0; j < 4; ++j) {
    const int ocol = colW + j * 16 + r;
    const float bv = (ocol < NFULL) ? bias[ocol] : 0.0f;
#pragma unroll
    for (int rr = 0; rr < 8; ++rr) {
      const int orow = rowBase + hi * 8 + rr;
      Cout[(size_t)orow * NPAD + ocol] = acc[j][rr] + bv;
    }
  }
}

// ---------------- layer 0: h0 = tanh(gene_out[:, :T0] @ W0^T + b0) ----------------
__global__ void layer0_fwd(const float* __restrict__ go, const float* __restrict__ W0,
                           const float* __restrict__ b0, float* __restrict__ h0) {
  int idx = blockIdx.x * blockDim.x + threadIdx.x;   // b*T0 + t
  if (idx >= BATCH * T0N) return;
  int b = idx / T0N, t = idx - b * T0N;
  const float* gi = go + (size_t)b * NPAD + t * KGN;
  float x[KGN];
#pragma unroll
  for (int k = 0; k < KGN; ++k) x[k] = gi[k];
  const float* w = W0 + (size_t)t * HN * KGN;
  float* ho = h0 + (size_t)idx * HN;
#pragma unroll
  for (int h = 0; h < HN; ++h) {
    float s = b0[t * HN + h];
#pragma unroll
    for (int k = 0; k < KGN; ++k) s += w[h * KGN + k] * x[k];
    ho[h] = tanhf(s);
  }
}

// ---------------- BatchNorm statistics: one block per channel ----------------
__global__ void bn_stats(const float* __restrict__ h, int nch,
                         float* __restrict__ mu, float* __restrict__ rv) {
  __shared__ float ssum[256];
  __shared__ float ssq[256];
  const int ch = blockIdx.x;
  float s = 0.f, q = 0.f;
  for (int b = threadIdx.x; b < BATCH; b += blockDim.x) {
    float v = h[(size_t)b * nch + ch];
    s += v; q += v * v;
  }
  ssum[threadIdx.x] = s; ssq[threadIdx.x] = q;
  __syncthreads();
  for (int st = 128; st > 0; st >>= 1) {
    if (threadIdx.x < st) {
      ssum[threadIdx.x] += ssum[threadIdx.x + st];
      ssq[threadIdx.x]  += ssq[threadIdx.x + st];
    }
    __syncthreads();
  }
  if (threadIdx.x == 0) {
    float m = ssum[0] * (1.0f / BATCH);
    float var = ssq[0] * (1.0f / BATCH) - m * m;   // biased, matches jnp.var
    mu[ch] = m;
    rv[ch] = rsqrtf(var + 1e-5f);
  }
}

// ---------------- BN apply (in place) + aux head, per (b, term) ----------------
__global__ void bn_aux(float* __restrict__ hbuf, int T,
                       const float* __restrict__ g, const float* __restrict__ be,
                       const float* __restrict__ mu, const float* __restrict__ rv,
                       const float* __restrict__ aW, const float* __restrict__ ab,
                       const float* __restrict__ cW, const float* __restrict__ cb,
                       float* __restrict__ out, int outOff) {
  int idx = blockIdx.x * blockDim.x + threadIdx.x;
  if (idx >= BATCH * T) return;
  int b = idx / T, t = idx - b * T;
  float* e = hbuf + (size_t)idx * HN;
  float a = ab[t];
#pragma unroll
  for (int h = 0; h < HN; ++h) {
    int ch = t * HN + h;
    float v = g[ch] * (e[h] - mu[ch]) * rv[ch] + be[ch];
    e[h] = v;                     // hbuf becomes the BN'd embedding
    a += aW[ch] * v;
  }
  a = tanhf(a);
  out[(size_t)b * OUTW + outOff + t] = cW[t] * a + cb[t];
}

// ---------------- layer 1: concat(child e0, direct genes) -> tanh-linear ----------------
__global__ void layer1_fwd(const float* __restrict__ e0, const float* __restrict__ go,
                           const float* __restrict__ W1, const float* __restrict__ b1,
                           float* __restrict__ h1) {
  const int IN1 = CCH * HN + KGN;  // 64
  int idx = blockIdx.x * blockDim.x + threadIdx.x;   // b*T1 + t
  if (idx >= BATCH * T1N) return;
  int b = idx / T1N, t = idx - b * T1N;
  float in[CCH * HN + KGN];
  const float* ep = e0 + ((size_t)b * T0N + t * CCH) * HN;   // 8 contiguous leaves * 6
#pragma unroll
  for (int i = 0; i < CCH * HN; ++i) in[i] = ep[i];
  const float* gp = go + (size_t)b * NPAD + (T0N + t) * KGN;
#pragma unroll
  for (int k = 0; k < KGN; ++k) in[CCH * HN + k] = gp[k];
  const float* w = W1 + (size_t)t * HN * IN1;
  float* ho = h1 + (size_t)idx * HN;
#pragma unroll
  for (int h = 0; h < HN; ++h) {
    float s = b1[t * HN + h];
#pragma unroll
    for (int i = 0; i < IN1; ++i) s += w[h * IN1 + i] * in[i];
    ho[h] = tanhf(s);
  }
}

// ---------------- root: concat(e1 flat, last-term genes) -> tanh-linear ----------------
__global__ void root_fwd(const float* __restrict__ e1, const float* __restrict__ go,
                         const float* __restrict__ W2, const float* __restrict__ b2,
                         float* __restrict__ h2) {
  const int IN2 = T1N * HN + KGN;  // 400
  int b = blockIdx.x * blockDim.x + threadIdx.x;
  if (b >= BATCH) return;
  float s[HN];
#pragma unroll
  for (int h = 0; h < HN; ++h) s[h] = b2[h];
  const float* ep = e1 + (size_t)b * (T1N * HN);
  for (int i = 0; i < T1N * HN; ++i) {
    float v = ep[i];
#pragma unroll
    for (int h = 0; h < HN; ++h) s[h] += W2[h * IN2 + i] * v;
  }
  const float* gp = go + (size_t)b * NPAD + (T0N + T1N) * KGN;  // term index TG-1
#pragma unroll
  for (int k = 0; k < KGN; ++k) {
    float v = gp[k];
#pragma unroll
    for (int h = 0; h < HN; ++h) s[h] += W2[h * IN2 + T1N * HN + k] * v;
  }
#pragma unroll
  for (int h = 0; h < HN; ++h) h2[(size_t)b * HN + h] = tanhf(s[h]);
}

// ---------------- root BN + aux2 + final head ----------------
__global__ void final_head(const float* __restrict__ h2,
                           const float* __restrict__ g2, const float* __restrict__ be2,
                           const float* __restrict__ mu, const float* __restrict__ rv,
                           const float* __restrict__ aW2, const float* __restrict__ ab2,
                           const float* __restrict__ cW2, const float* __restrict__ cb2,
                           const float* __restrict__ Wf, const float* __restrict__ bfp,
                           const float* __restrict__ Wo, const float* __restrict__ bo,
                           float* __restrict__ out) {
  int b = blockIdx.x * blockDim.x + threadIdx.x;
  if (b >= BATCH) return;
  float e[HN];
#pragma unroll
  for (int h = 0; h < HN; ++h)
    e[h] = g2[h] * (h2[(size_t)b * HN + h] - mu[h]) * rv[h] + be2[h];
  float a = ab2[0];
#pragma unroll
  for (int h = 0; h < HN; ++h) a += aW2[h] * e[h];
  a = tanhf(a);
  out[(size_t)b * OUTW + T0N + T1N] = cW2[0] * a + cb2[0];
  float f = bfp[0];
#pragma unroll
  for (int h = 0; h < HN; ++h) f += Wf[h] * e[h];
  f = tanhf(f);
  out[(size_t)b * OUTW + T0N + T1N + 1] = Wo[0] * f + bo[0];
}

extern "C" void kernel_launch(void* const* d_in, const int* in_sizes, int n_in,
                              void* d_out, int out_size, void* d_ws, size_t ws_size,
                              hipStream_t stream) {
  // setup_inputs order: gene_input, then params dict in insertion order.
  const float* x   = (const float*)d_in[0];
  const float* Wg  = (const float*)d_in[1];
  const float* bg  = (const float*)d_in[2];
  const float* W0  = (const float*)d_in[3];
  const float* b0  = (const float*)d_in[4];
  const float* g0  = (const float*)d_in[5];
  const float* be0 = (const float*)d_in[6];
  const float* aW0 = (const float*)d_in[7];
  const float* ab0 = (const float*)d_in[8];
  const float* cW0 = (const float*)d_in[9];
  const float* cb0 = (const float*)d_in[10];
  const float* W1  = (const float*)d_in[11];
  const float* b1  = (const float*)d_in[12];
  const float* g1  = (const float*)d_in[13];
  const float* be1 = (const float*)d_in[14];
  const float* aW1 = (const float*)d_in[15];
  const float* ab1 = (const float*)d_in[16];
  const float* cW1 = (const float*)d_in[17];
  const float* cb1 = (const float*)d_in[18];
  const float* W2  = (const float*)d_in[19];
  const float* b2  = (const float*)d_in[20];
  const float* g2  = (const float*)d_in[21];
  const float* be2 = (const float*)d_in[22];
  const float* aW2 = (const float*)d_in[23];
  const float* ab2 = (const float*)d_in[24];
  const float* cW2 = (const float*)d_in[25];
  const float* cb2 = (const float*)d_in[26];
  const float* Wf  = (const float*)d_in[27];
  const float* bf_ = (const float*)d_in[28];
  const float* Wo  = (const float*)d_in[29];
  const float* bo  = (const float*)d_in[30];
  float* out = (float*)d_out;
  (void)in_sizes; (void)n_in; (void)out_size; (void)ws_size;

  // ---- workspace carving (256-B aligned) ----
  char* ws = (char*)d_ws;
  size_t off = 0;
  auto take = [&](size_t bytes) -> char* {
    char* p = ws + off;
    off += (bytes + 255) & ~(size_t)255;
    return p;
  };
  unsigned short* xb = (unsigned short*)take((size_t)BATCH * GDIM * 2);     // x in bf16
  unsigned short* wb = (unsigned short*)take((size_t)NPAD  * GDIM * 2);     // Wg in bf16, padded
  float* go = (float*)take((size_t)BATCH * NPAD * 4);                       // gene_out
  float* h0 = (float*)take((size_t)BATCH * T0N * HN * 4);                   // h0 -> e0 (in place)
  float* h1 = (float*)take((size_t)BATCH * T1N * HN * 4);                   // h1 -> e1 (in place)
  float* h2 = (float*)take((size_t)BATCH * HN * 4);
  float* s0 = (float*)take((size_t)2 * T0N * HN * 4);                       // mu | rvar
  float* s1 = (float*)take((size_t)2 * T1N * HN * 4);
  float* s2 = (float*)take((size_t)2 * HN * 4);

  // 1) precision conversion
  {
    long long tot = (long long)BATCH * GDIM;
    cvt_bf16_pad<<<(unsigned)((tot + 255) / 256), 256, 0, stream>>>(x, xb, BATCH, BATCH, GDIM);
  }
  {
    long long tot = (long long)NPAD * GDIM;
    cvt_bf16_pad<<<(unsigned)((tot + 255) / 256), 256, 0, stream>>>(Wg, wb, NFULL, NPAD, GDIM);
  }

  // 2) the dominant GEMM on the WMMA + async-LDS path
  dim3 gg(NPAD / 128, BATCH / 64);
  gemm_gene_wmma<<<gg, 256, 0, stream>>>(xb, wb, bg, go);

  // 3) leaves
  layer0_fwd<<<(BATCH * T0N + 255) / 256, 256, 0, stream>>>(go, W0, b0, h0);
  bn_stats<<<T0N * HN, 256, 0, stream>>>(h0, T0N * HN, s0, s0 + T0N * HN);
  bn_aux<<<(BATCH * T0N + 255) / 256, 256, 0, stream>>>(h0, T0N, g0, be0, s0, s0 + T0N * HN,
                                                        aW0, ab0, cW0, cb0, out, 0);

  // 4) parents
  layer1_fwd<<<(BATCH * T1N + 255) / 256, 256, 0, stream>>>(h0, go, W1, b1, h1);
  bn_stats<<<T1N * HN, 256, 0, stream>>>(h1, T1N * HN, s1, s1 + T1N * HN);
  bn_aux<<<(BATCH * T1N + 255) / 256, 256, 0, stream>>>(h1, T1N, g1, be1, s1, s1 + T1N * HN,
                                                        aW1, ab1, cW1, cb1, out, T0N);

  // 5) root + final head
  root_fwd<<<(BATCH + 255) / 256, 256, 0, stream>>>(h1, go, W2, b2, h2);
  bn_stats<<<HN, 256, 0, stream>>>(h2, HN, s2, s2 + HN);
  final_head<<<(BATCH + 255) / 256, 256, 0, stream>>>(h2, g2, be2, s2, s2 + HN,
                                                      aW2, ab2, cW2, cb2,
                                                      Wf, bf_, Wo, bo, out);
}
